// AttnBlock_17995912970351
// MI455X (gfx1250) — matrix operand order repaired and checked by reading, not compiled, not run on hardware
//
#include <hip/hip_runtime.h>
#include <hip/hip_bf16.h>

// ---------------------------------------------------------------------------
// AttnBlock on MI455X (gfx1250).  All four GEMM stages on v_wmma_f32_16x16x32_f16.
// K-loops are software-pipelined with even/odd register buffers (no cur<-nxt
// copies, sched_barrier-fenced) so loads for step k+1 issue before the WMMAs of
// step k.  Scores kernel stages its 32-row q-panel in LDS via the CDNA5
// async-to-LDS path (global_load_async_to_lds_b128 + s_wait_asynccnt).
// B=2, C=512, HW=4096, 32 groups.
// ---------------------------------------------------------------------------

typedef __attribute__((ext_vector_type(16))) _Float16 v16h;
typedef __attribute__((ext_vector_type(8)))  _Float16 v8h;
typedef __attribute__((ext_vector_type(8)))  float    v8f;

#define HW   4096
#define CCH  512
#define NB   2

#define SCHED_FENCE() __builtin_amdgcn_sched_barrier(0)

// ---- WMMA operand loaders (ISA 7.12.2 layouts, wave32) ---------------------
__device__ __forceinline__ v16h load_A16x32(const _Float16* __restrict__ p,
                                            int ld, int lane) {
  const _Float16* row = p + (size_t)(lane & 15) * (size_t)ld;
  const int kc = (lane >> 4) * 8;
  v8h lo = *(const v8h*)(row + kc);
  v8h hh = *(const v8h*)(row + 16 + kc);
  v16h r;
#pragma unroll
  for (int t = 0; t < 8; ++t) { r[t] = lo[t]; r[t + 8] = hh[t]; }
  return r;
}

__device__ __forceinline__ v16h load_B32x16(const _Float16* __restrict__ p,
                                            int ld, int lane) {
  const _Float16* col = p + (size_t)(lane & 15) * (size_t)ld;
  const int kc = (lane >> 4) * 8;
  v8h lo = *(const v8h*)(col + kc);
  v8h hh = *(const v8h*)(col + 16 + kc);
  v16h r;
#pragma unroll
  for (int t = 0; t < 8; ++t) { r[t] = lo[t]; r[t + 8] = hh[t]; }
  return r;
}

__device__ __forceinline__ void atomicMaxF(float* addr, float val) {
  unsigned int* ua = (unsigned int*)addr;
  unsigned int old = __float_as_uint(*addr);
  while (val > __uint_as_float(old)) {
    unsigned int assumed = old;
    old = atomicCAS(ua, assumed, __float_as_uint(val));
    if (old == assumed) break;
  }
}

#define WMMA_F16(A, B, C) \
  __builtin_amdgcn_wmma_f32_16x16x32_f16(false, (A), false, (B), (short)0, (C), false, false)

// ---- kernel 0: weights -> f16, init rowmax --------------------------------
__global__ void prep_kernel(const float* __restrict__ wq, const float* __restrict__ wk,
                            const float* __restrict__ wv, const float* __restrict__ wo,
                            _Float16* wqh, _Float16* wkh, _Float16* wvh, _Float16* woh,
                            float* mbuf) {
  int idx = blockIdx.x * 256 + threadIdx.x;
  const int CC = CCH * CCH;
  if (idx < CC) {
    wqh[idx] = (_Float16)wq[idx];
    wkh[idx] = (_Float16)wk[idx];
    wvh[idx] = (_Float16)wv[idx];
    woh[idx] = (_Float16)wo[idx];
  }
  if (idx < NB * HW) mbuf[idx] = -3.0e38f;
}

// ---- kernel 1: GroupNorm -> hnT (f16, i-major) -----------------------------
__global__ void gn_kernel(const float* __restrict__ x, const float* __restrict__ gamma,
                          const float* __restrict__ beta, _Float16* __restrict__ hnT) {
  const int b = blockIdx.x >> 5;
  const int g = blockIdx.x & 31;
  const size_t base = ((size_t)b * CCH + g * 16) * HW;
  const int tid = threadIdx.x;
  float s = 0.f, s2 = 0.f;
  for (int idx = tid; idx < 16 * HW; idx += 256) {
    float v = x[base + idx];
    s += v; s2 += v * v;
  }
  __shared__ float sh1[256], sh2[256];
  sh1[tid] = s; sh2[tid] = s2;
  __syncthreads();
  for (int off = 128; off > 0; off >>= 1) {
    if (tid < off) { sh1[tid] += sh1[tid + off]; sh2[tid] += sh2[tid + off]; }
    __syncthreads();
  }
  const float mean = sh1[0] * (1.f / 65536.f);
  const float var  = sh2[0] * (1.f / 65536.f) - mean * mean;
  const float rstd = rsqrtf(var + 1e-6f);
  for (int idx = tid; idx < 16 * HW; idx += 256) {
    int cl = idx >> 12;
    int i  = idx & (HW - 1);
    int c  = g * 16 + cl;
    float v = (x[base + idx] - mean) * rstd * gamma[c] + beta[c];
    hnT[((size_t)b * HW + i) * CCH + c] = (_Float16)v;
  }
}

// ---- kernel 2: Q,K projection (D: M=i, N=o, 16x32/wave), even/odd pipeline -
__global__ void proj_qk_kernel(const _Float16* __restrict__ hnT,
                               const _Float16* __restrict__ wqh,
                               const _Float16* __restrict__ wkh,
                               const float* __restrict__ bq, const float* __restrict__ bk,
                               _Float16* __restrict__ qT, _Float16* __restrict__ kT) {
  const int lane = threadIdx.x & 31;
  const int wid  = blockIdx.x * 8 + (threadIdx.x >> 5);  // 8192 waves
  const int oc = wid & 15, it = (wid >> 4) & 255, b = wid >> 12;
  const _Float16* Abase = hnT + ((size_t)b * HW + it * 16) * CCH;
  const _Float16* Wq0 = wqh + (size_t)(oc * 32) * CCH;
  const _Float16* Wk0 = wkh + (size_t)(oc * 32) * CCH;
  v8f accq[2] = {}, acck[2] = {};
  v16h a_ev, a_od, bq_ev[2], bk_ev[2], bq_od[2], bk_od[2];
  a_ev = load_A16x32(Abase, CCH, lane);
#pragma unroll
  for (int s = 0; s < 2; ++s) {
    bq_ev[s] = load_B32x16(Wq0 + (size_t)(s * 16) * CCH, CCH, lane);
    bk_ev[s] = load_B32x16(Wk0 + (size_t)(s * 16) * CCH, CCH, lane);
  }
  for (int k0 = 0; k0 < CCH; k0 += 64) {
    a_od = load_A16x32(Abase + k0 + 32, CCH, lane);
#pragma unroll
    for (int s = 0; s < 2; ++s) {
      bq_od[s] = load_B32x16(Wq0 + (size_t)(s * 16) * CCH + k0 + 32, CCH, lane);
      bk_od[s] = load_B32x16(Wk0 + (size_t)(s * 16) * CCH + k0 + 32, CCH, lane);
    }
    SCHED_FENCE();
#pragma unroll
    for (int s = 0; s < 2; ++s) {
      accq[s] = WMMA_F16(a_ev, bq_ev[s], accq[s]);
      acck[s] = WMMA_F16(a_ev, bk_ev[s], acck[s]);
    }
    SCHED_FENCE();
    if (k0 + 64 < CCH) {
      a_ev = load_A16x32(Abase + k0 + 64, CCH, lane);
#pragma unroll
      for (int s = 0; s < 2; ++s) {
        bq_ev[s] = load_B32x16(Wq0 + (size_t)(s * 16) * CCH + k0 + 64, CCH, lane);
        bk_ev[s] = load_B32x16(Wk0 + (size_t)(s * 16) * CCH + k0 + 64, CCH, lane);
      }
    }
    SCHED_FENCE();
#pragma unroll
    for (int s = 0; s < 2; ++s) {
      accq[s] = WMMA_F16(a_od, bq_od[s], accq[s]);
      acck[s] = WMMA_F16(a_od, bk_od[s], acck[s]);
    }
    SCHED_FENCE();
  }
  const float scale = 0.04419417382415922f;  // 512^-0.5
  const int hi = lane >> 4, n = lane & 15;
#pragma unroll
  for (int s = 0; s < 2; ++s) {
    int o = oc * 32 + s * 16 + n;
    float vbq = bq[o], vbk = bk[o];
#pragma unroll
    for (int r = 0; r < 8; ++r) {
      int M = r + 8 * hi;
      size_t idx = ((size_t)b * HW + it * 16 + M) * CCH + o;
      qT[idx] = (_Float16)((accq[s][r] + vbq) * scale);
      kT[idx] = (_Float16)(acck[s][r] + vbk);
    }
  }
}

// ---- kernel 3: V projection (D: M=c, N=j, 16x64/wave), even/odd pipeline ---
__global__ void proj_v_kernel(const _Float16* __restrict__ hnT,
                              const _Float16* __restrict__ wvh,
                              const float* __restrict__ bv,
                              _Float16* __restrict__ vC) {
  const int lane = threadIdx.x & 31;
  const int wid  = blockIdx.x * 8 + (threadIdx.x >> 5);  // 4096 waves
  const int jc = wid & 63, ct = (wid >> 6) & 31, b = wid >> 11;
  const _Float16* Abase = wvh + (size_t)(ct * 16) * CCH;
  const _Float16* Bb[4];
#pragma unroll
  for (int s = 0; s < 4; ++s)
    Bb[s] = hnT + ((size_t)b * HW + jc * 64 + s * 16) * CCH;
  v8f acc[4] = {};
  v16h a_ev, a_od, b_ev[4], b_od[4];
  a_ev = load_A16x32(Abase, CCH, lane);
#pragma unroll
  for (int s = 0; s < 4; ++s) b_ev[s] = load_B32x16(Bb[s], CCH, lane);
  for (int k0 = 0; k0 < CCH; k0 += 64) {
    a_od = load_A16x32(Abase + k0 + 32, CCH, lane);
#pragma unroll
    for (int s = 0; s < 4; ++s) b_od[s] = load_B32x16(Bb[s] + k0 + 32, CCH, lane);
    SCHED_FENCE();
#pragma unroll
    for (int s = 0; s < 4; ++s) acc[s] = WMMA_F16(a_ev, b_ev[s], acc[s]);
    SCHED_FENCE();
    if (k0 + 64 < CCH) {
      a_ev = load_A16x32(Abase + k0 + 64, CCH, lane);
#pragma unroll
      for (int s = 0; s < 4; ++s) b_ev[s] = load_B32x16(Bb[s] + k0 + 64, CCH, lane);
    }
    SCHED_FENCE();
#pragma unroll
    for (int s = 0; s < 4; ++s) acc[s] = WMMA_F16(a_od, b_od[s], acc[s]);
    SCHED_FENCE();
  }
  const int hi = lane >> 4, n = lane & 15;
#pragma unroll
  for (int r = 0; r < 8; ++r) {
    int M = r + 8 * hi;
    float vb = bv[ct * 16 + M];
#pragma unroll
    for (int s = 0; s < 4; ++s) {
      vC[((size_t)b * CCH + ct * 16 + M) * HW + jc * 64 + s * 16 + n] =
          (_Float16)(acc[s][r] + vb);
    }
  }
}

// ---- kernel 4: scores S = Q^T K (M=32, N=64 per wave).  Block stages its
// 32-row q-panel (32 KiB) in LDS via async-to-LDS; each kT B-tile then feeds
// two WMMAs (halved B traffic per WMMA).
__global__ void scores_kernel(const _Float16* __restrict__ qT,
                              const _Float16* __restrict__ kT,
                              _Float16* __restrict__ Sbuf, float* __restrict__ mbuf) {
  const int lane = threadIdx.x & 31;
  const int blk  = blockIdx.x;                      // 2048 blocks
  const int it2 = (blk >> 3) & 127, b = blk >> 10;  // block-uniform
  const int jc  = (blk & 7) * 8 + (threadIdx.x >> 5);

  __shared__ _Float16 sA[32 * CCH];                 // 32 KiB q-panel
  {
    const _Float16* gsrc = qT + ((size_t)b * HW + it2 * 32) * CCH;
    const int t = threadIdx.x;
#pragma unroll
    for (int c = 0; c < 8; ++c) {
      int chunk = t + 256 * c;                      // 2048 x 16B transfers
      unsigned loff = (unsigned)(size_t)&sA[chunk * 8];
      unsigned long long gaddr = (unsigned long long)(size_t)(gsrc + chunk * 8);
      asm volatile("global_load_async_to_lds_b128 %0, %1, off"
                   :: "v"(loff), "v"(gaddr) : "memory");
    }
    asm volatile("s_wait_asynccnt 0x0" ::: "memory");
    __syncthreads();
  }

  const _Float16* Bb[4];
#pragma unroll
  for (int s = 0; s < 4; ++s)
    Bb[s] = kT + ((size_t)b * HW + jc * 64 + s * 16) * CCH;

  const int row0 = (lane & 15) * CCH;
  const int kc   = (lane >> 4) * 8;
  auto ldsA = [&](int rbase, int k0) -> v16h {
    v8h lo = *(const v8h*)&sA[rbase + k0 + kc];
    v8h hh = *(const v8h*)&sA[rbase + k0 + 16 + kc];
    v16h a;
#pragma unroll
    for (int t = 0; t < 8; ++t) { a[t] = lo[t]; a[t + 8] = hh[t]; }
    return a;
  };

  v8f acc[2][4] = {};
  v16h b_ev[4], b_od[4];
#pragma unroll
  for (int s = 0; s < 4; ++s) b_ev[s] = load_B32x16(Bb[s], CCH, lane);

  for (int k0 = 0; k0 < CCH; k0 += 64) {
#pragma unroll
    for (int s = 0; s < 4; ++s) b_od[s] = load_B32x16(Bb[s] + k0 + 32, CCH, lane);
    SCHED_FENCE();
    {
      v16h a0 = ldsA(row0, k0);
      v16h a1 = ldsA(row0 + 16 * CCH, k0);
#pragma unroll
      for (int s = 0; s < 4; ++s) {
        acc[0][s] = WMMA_F16(a0, b_ev[s], acc[0][s]);
        acc[1][s] = WMMA_F16(a1, b_ev[s], acc[1][s]);
      }
    }
    SCHED_FENCE();
    if (k0 + 64 < CCH) {
#pragma unroll
      for (int s = 0; s < 4; ++s) b_ev[s] = load_B32x16(Bb[s] + k0 + 64, CCH, lane);
    }
    SCHED_FENCE();
    {
      v16h a0 = ldsA(row0, k0 + 32);
      v16h a1 = ldsA(row0 + 16 * CCH, k0 + 32);
#pragma unroll
      for (int s = 0; s < 4; ++s) {
        acc[0][s] = WMMA_F16(a0, b_od[s], acc[0][s]);
        acc[1][s] = WMMA_F16(a1, b_od[s], acc[1][s]);
      }
    }
    SCHED_FENCE();
  }

  const int hi = lane >> 4, n = lane & 15;
#pragma unroll
  for (int m = 0; m < 2; ++m) {
#pragma unroll
    for (int r = 0; r < 8; ++r) {
      int M = m * 16 + r + 8 * hi;
      float mv = -3.0e38f;
#pragma unroll
      for (int s = 0; s < 4; ++s) mv = fmaxf(mv, acc[m][s][r]);
      mv = fmaxf(mv, __shfl_xor(mv, 1, 32));
      mv = fmaxf(mv, __shfl_xor(mv, 2, 32));
      mv = fmaxf(mv, __shfl_xor(mv, 4, 32));
      mv = fmaxf(mv, __shfl_xor(mv, 8, 32));
      if (n == 0) atomicMaxF(&mbuf[(size_t)b * HW + it2 * 32 + M], mv);
#pragma unroll
      for (int s = 0; s < 4; ++s) {
        Sbuf[((size_t)b * HW + it2 * 32 + M) * HW + jc * 64 + s * 16 + n] =
            (_Float16)acc[m][s][r];
      }
    }
  }
}

// ---- kernel 5: fused softmax + P*V^T (M=32, N=64 per wave), even/odd -------
__global__ void pv_kernel(const _Float16* __restrict__ Sbuf,
                          const float* __restrict__ mbuf,
                          const _Float16* __restrict__ vC,
                          _Float16* __restrict__ oT) {
  const int lane = threadIdx.x & 31;
  const int wid  = blockIdx.x * 8 + (threadIdx.x >> 5);  // 2048 waves
  const int cc = wid & 7, it2 = (wid >> 3) & 127, b = wid >> 10;
  const int rlo = lane & 15;
  const float m0 = mbuf[(size_t)b * HW + it2 * 32 + rlo];
  const float m1 = mbuf[(size_t)b * HW + it2 * 32 + 16 + rlo];
  const _Float16* S0 = Sbuf + ((size_t)b * HW + it2 * 32) * HW;
  const _Float16* S1 = S0 + (size_t)16 * HW;
  const _Float16* Bb[4];
#pragma unroll
  for (int s = 0; s < 4; ++s)
    Bb[s] = vC + ((size_t)b * CCH + cc * 64 + s * 16) * HW;

  v8f acc[2][4] = {};
  float ls0 = 0.f, ls1 = 0.f;
  v16h b_ev[4], b_od[4];
#pragma unroll
  for (int s = 0; s < 4; ++s) b_ev[s] = load_B32x16(Bb[s], HW, lane);

  for (int jb = 0; jb < HW; jb += 64) {
#pragma unroll
    for (int s = 0; s < 4; ++s) b_od[s] = load_B32x16(Bb[s] + jb + 32, HW, lane);
    __builtin_prefetch(S0 + jb + 64, 0, 1);
    __builtin_prefetch(S1 + jb + 64, 0, 1);
    SCHED_FENCE();
    {
      v16h sa0 = load_A16x32(S0 + jb, HW, lane);
      v16h sa1 = load_A16x32(S1 + jb, HW, lane);
      v16h p0, p1;
      float pt0 = 0.f, pt1 = 0.f;
#pragma unroll
      for (int t = 0; t < 16; ++t) {
        float e0 = __expf((float)sa0[t] - m0);
        float e1 = __expf((float)sa1[t] - m1);
        p0[t] = (_Float16)e0; p1[t] = (_Float16)e1;
        pt0 += e0; pt1 += e1;
      }
      ls0 += pt0; ls1 += pt1;
#pragma unroll
      for (int s = 0; s < 4; ++s) {
        acc[0][s] = WMMA_F16(p0, b_ev[s], acc[0][s]);
        acc[1][s] = WMMA_F16(p1, b_ev[s], acc[1][s]);
      }
    }
    SCHED_FENCE();
    if (jb + 64 < HW) {
#pragma unroll
      for (int s = 0; s < 4; ++s) b_ev[s] = load_B32x16(Bb[s] + jb + 64, HW, lane);
    }
    SCHED_FENCE();
    {
      v16h sa0 = load_A16x32(S0 + jb + 32, HW, lane);
      v16h sa1 = load_A16x32(S1 + jb + 32, HW, lane);
      v16h p0, p1;
      float pt0 = 0.f, pt1 = 0.f;
#pragma unroll
      for (int t = 0; t < 16; ++t) {
        float e0 = __expf((float)sa0[t] - m0);
        float e1 = __expf((float)sa1[t] - m1);
        p0[t] = (_Float16)e0; p1[t] = (_Float16)e1;
        pt0 += e0; pt1 += e1;
      }
      ls0 += pt0; ls1 += pt1;
#pragma unroll
      for (int s = 0; s < 4; ++s) {
        acc[0][s] = WMMA_F16(p0, b_od[s], acc[0][s]);
        acc[1][s] = WMMA_F16(p1, b_od[s], acc[1][s]);
      }
    }
    SCHED_FENCE();
  }
  ls0 += __shfl_xor(ls0, 16, 32);     // combine the two K-half partials
  ls1 += __shfl_xor(ls1, 16, 32);
  float rl0[16], rl1[16];
#pragma unroll
  for (int r = 0; r < 16; ++r) {
    rl0[r] = __uint_as_float(__builtin_amdgcn_readlane(__float_as_uint(ls0), r));
    rl1[r] = __uint_as_float(__builtin_amdgcn_readlane(__float_as_uint(ls1), r));
  }
  const int hi = lane >> 4, n = lane & 15;
#pragma unroll
  for (int m = 0; m < 2; ++m) {
#pragma unroll
    for (int r = 0; r < 8; ++r) {
      int M = m * 16 + r + 8 * hi;
      float l  = m ? (hi ? rl1[r + 8] : rl1[r]) : (hi ? rl0[r + 8] : rl0[r]);
      float inv = 1.f / l;
#pragma unroll
      for (int s = 0; s < 4; ++s) {
        oT[((size_t)b * HW + it2 * 32 + M) * CCH + cc * 64 + s * 16 + n] =
            (_Float16)(acc[m][s][r] * inv);
      }
    }
  }
}

// ---- kernel 6: output projection + bias + residual (D: M=o, N=i), even/odd -
__global__ void outproj_kernel(const _Float16* __restrict__ oT,
                               const _Float16* __restrict__ woh,
                               const float* __restrict__ bo,
                               const float* __restrict__ x,
                               float* __restrict__ out) {
  const int lane = threadIdx.x & 31;
  const int wid  = blockIdx.x * 8 + (threadIdx.x >> 5);  // 4096 waves
  const int ic = wid & 63, ot = (wid >> 6) & 31, b = wid >> 11;
  const _Float16* Abase = woh + (size_t)(ot * 16) * CCH;
  const _Float16* Bb[4];
#pragma unroll
  for (int s = 0; s < 4; ++s)
    Bb[s] = oT + ((size_t)b * HW + ic * 64 + s * 16) * CCH;
  v8f acc[4] = {};
  v16h a_ev, a_od, b_ev[4], b_od[4];
  a_ev = load_A16x32(Abase, CCH, lane);
#pragma unroll
  for (int s = 0; s < 4; ++s) b_ev[s] = load_B32x16(Bb[s], CCH, lane);
  for (int k0 = 0; k0 < CCH; k0 += 64) {
    a_od = load_A16x32(Abase + k0 + 32, CCH, lane);
#pragma unroll
    for (int s = 0; s < 4; ++s) b_od[s] = load_B32x16(Bb[s] + k0 + 32, CCH, lane);
    SCHED_FENCE();
#pragma unroll
    for (int s = 0; s < 4; ++s) acc[s] = WMMA_F16(a_ev, b_ev[s], acc[s]);
    SCHED_FENCE();
    if (k0 + 64 < CCH) {
      a_ev = load_A16x32(Abase + k0 + 64, CCH, lane);
#pragma unroll
      for (int s = 0; s < 4; ++s) b_ev[s] = load_B32x16(Bb[s] + k0 + 64, CCH, lane);
    }
    SCHED_FENCE();
#pragma unroll
    for (int s = 0; s < 4; ++s) acc[s] = WMMA_F16(a_od, b_od[s], acc[s]);
    SCHED_FENCE();
  }
  const int hi = lane >> 4, n = lane & 15;
#pragma unroll
  for (int r = 0; r < 8; ++r) {
    int M = r + 8 * hi;
    float vb = bo[ot * 16 + M];
#pragma unroll
    for (int s = 0; s < 4; ++s) {
      size_t idx = ((size_t)b * CCH + ot * 16 + M) * HW + ic * 64 + s * 16 + n;
      out[idx] = x[idx] + acc[s][r] + vb;
    }
  }
}

// ---------------------------------------------------------------------------
extern "C" void kernel_launch(void* const* d_in, const int* in_sizes, int n_in,
                              void* d_out, int out_size, void* d_ws, size_t ws_size,
                              hipStream_t stream) {
  (void)in_sizes; (void)n_in; (void)out_size; (void)ws_size;
  const float* x     = (const float*)d_in[0];
  const float* gamma = (const float*)d_in[1];
  const float* beta  = (const float*)d_in[2];
  const float* wq = (const float*)d_in[3];  const float* bq = (const float*)d_in[4];
  const float* wk = (const float*)d_in[5];  const float* bk = (const float*)d_in[6];
  const float* wv = (const float*)d_in[7];  const float* bv = (const float*)d_in[8];
  const float* wo = (const float*)d_in[9];  const float* bo = (const float*)d_in[10];
  float* out = (float*)d_out;

  // workspace partition (~106 MiB total)
  char* ws = (char*)d_ws;
  const size_t szT = (size_t)NB * HW * CCH * sizeof(_Float16);   // 8 MiB each
  const size_t szW = (size_t)CCH * CCH * sizeof(_Float16);       // 512 KiB each
  const size_t szM = (size_t)NB * HW * sizeof(float);            // 32 KiB
  _Float16* hnT  = (_Float16*)(ws);                   size_t off = szT;
  _Float16* qT   = (_Float16*)(ws + off); off += szT;
  _Float16* kT   = (_Float16*)(ws + off); off += szT;
  _Float16* vC   = (_Float16*)(ws + off); off += szT;
  _Float16* oT   = (_Float16*)(ws + off); off += szT;
  _Float16* wqh  = (_Float16*)(ws + off); off += szW;
  _Float16* wkh  = (_Float16*)(ws + off); off += szW;
  _Float16* wvh  = (_Float16*)(ws + off); off += szW;
  _Float16* woh  = (_Float16*)(ws + off); off += szW;
  float*    mbuf = (float*)   (ws + off); off += szM;
  _Float16* Sbuf = (_Float16*)(ws + off);             // 64 MiB

  prep_kernel   <<<1024, 256, 0, stream>>>(wq, wk, wv, wo, wqh, wkh, wvh, woh, mbuf);
  gn_kernel     <<<  64, 256, 0, stream>>>(x, gamma, beta, hnT);
  proj_qk_kernel<<<1024, 256, 0, stream>>>(hnT, wqh, wkh, bq, bk, qT, kT);
  proj_v_kernel <<< 512, 256, 0, stream>>>(hnT, wvh, bv, vC);
  scores_kernel <<<2048, 256, 0, stream>>>(qT, kT, Sbuf, mbuf);
  pv_kernel     <<< 256, 256, 0, stream>>>(Sbuf, mbuf, vC, oT);
  outproj_kernel<<< 512, 256, 0, stream>>>(oT, woh, bo, x, out);
}
